// SimpleDAGNN_86466281603216
// MI455X (gfx1250) — matrix operand order repaired and checked by reading, not compiled, not run on hardware
//
#include <hip/hip_runtime.h>
#include <stddef.h>
#include <stdint.h>

#define NN   50000      // nodes
#define NE   800000     // edges
#define DIN  1536
#define HF   256        // hidden
#define NG   64         // graphs

typedef __attribute__((ext_vector_type(16))) __bf16 v16bf;
typedef __attribute__((ext_vector_type(8)))  float  v8f;
typedef unsigned short     u16;
typedef unsigned int       u32;
typedef unsigned long long u64;

__device__ __forceinline__ u16 f32_to_bf16(float f) {
    union { __bf16 b; u16 u; } c;
    c.b = (__bf16)f;                       // RNE, hardware cvt when available
    return c.u;
}
__device__ __forceinline__ float bf16_to_f32(u16 h) {
    return __uint_as_float(((u32)h) << 16);
}

// CDNA5 async DMA: global -> LDS, 16B per lane, tracked by ASYNCcnt
__device__ __forceinline__ void async_b128(u32 lds_addr, u32 voff, u64 sbase) {
    asm volatile("global_load_async_to_lds_b128 %0, %1, %2"
                 :: "v"(lds_addr), "v"(voff), "s"(sbase) : "memory");
}
__device__ __forceinline__ void wait_async0() {
    asm volatile("s_wait_asynccnt 0x0" ::: "memory");
}

// ---------------- small utility kernels ----------------

__global__ void fill_f32(float* __restrict__ p, float v, int n) {
    int i = blockIdx.x * blockDim.x + threadIdx.x;
    if (i < n) p[i] = v;
}

__global__ void deg_accum(const int* __restrict__ dst, float* __restrict__ deg, int e) {
    int i = blockIdx.x * blockDim.x + threadIdx.x;
    if (i < e) atomicAdd(&deg[dst[i]], 1.0f);
}

__global__ void deg_to_dinv(float* __restrict__ d, int n) {
    int i = blockIdx.x * blockDim.x + threadIdx.x;
    if (i < n) { float v = d[i]; d[i] = (v > 0.0f) ? rsqrtf(v) : 0.0f; }
}

// Wt[n*K + k] = bf16(W[k*Nn + n])   (convert + transpose weights)
__global__ void cvt_transpose(const float* __restrict__ W, u16* __restrict__ Wt,
                              int K, int Nn) {
    int i = blockIdx.x * blockDim.x + threadIdx.x;
    if (i < K * Nn) {
        int k = i / Nn, n = i - k * Nn;
        Wt[(size_t)n * K + k] = f32_to_bf16(W[i]);
    }
}

// ---------------- WMMA GEMM: C[M,256] = A[M,K] @ B[K,256] (+bias) ----------------
// A: fp32 (guarded, converted in VALU) or bf16 (async DMA to LDS, unguarded
//    over-read stays inside workspace). Bt: bf16 transposed [256][K].
// Block: 256 threads (8 waves). Tile: 64 M x 256 N, K-step 32, double-buffered LDS,
// async-to-LDS staging, batched fragment loads so dscnt waits are partial.

template<bool A_F32, bool OUT_BF16, bool ADD_BIAS>
__global__ __launch_bounds__(256) void gemm_wmma(
    const void* __restrict__ Av, const u16* __restrict__ Bt,
    const float* __restrict__ bias, void* __restrict__ Out,
    int M, int K)
{
    constexpr int AP  = 40;                // padded LDS row (bf16), 80B: keeps 16B align
    constexpr int ASZ = 64 * AP;
    constexpr int BSZ = 256 * AP;
    __shared__ u16 As[2 * ASZ];            // 2 x 5.0 KB
    __shared__ u16 Bs[2 * BSZ];            // 2 x 20.0 KB

    const int tid   = threadIdx.x;
    const int lane  = tid & 31;
    const int wave  = tid >> 5;
    const int mbase = blockIdx.x * 64;

    const int half = lane >> 4;            // lane half (ISA 16-bit A/B layout)
    const int l15  = lane & 15;

    const int mstrip = wave & 3;           // which 16-row strip of the 64
    const int nbase  = (wave >> 2) * 128;  // cols [nbase, nbase+128)

    const int lrow   = tid >> 2;           // 0..63: A row to stage
    const int lchunk = tid & 3;            // 8 elems per chunk

    v8f acc[8];
    #pragma unroll
    for (int c = 0; c < 8; ++c)
        #pragma unroll
        for (int r = 0; r < 8; ++r) acc[c][r] = 0.0f;

    const int S = K >> 5;                  // K-steps of 32

    auto stage = [&](int ks, int bufi) {
        const int k0 = ks * 32;
        u16* Asb = As + bufi * ASZ;
        u16* Bsb = Bs + bufi * BSZ;
        // ---- B tile (256 n x 32 k): 4 async b128 per thread ----
        {
            u32 lds  = (u32)(uintptr_t)&Bsb[tid * AP];
            u32 voff = (u32)(tid * K) * 2u;
            u64 base = (u64)(uintptr_t)(Bt + k0);
            #pragma unroll
            for (int c = 0; c < 4; ++c)
                async_b128(lds + c * 16, voff + c * 16, base);
        }
        // ---- A tile (64 x 32) ----
        if (A_F32) {
            union { u16 h[8]; uint4 q; } t;
            int gm = mbase + lrow;
            if (gm < M) {
                const float* A = (const float*)Av;
                const float4* s4 = (const float4*)(A + (size_t)gm * K + k0 + lchunk * 8);
                float4 f0 = s4[0], f1 = s4[1];
                t.h[0] = f32_to_bf16(f0.x); t.h[1] = f32_to_bf16(f0.y);
                t.h[2] = f32_to_bf16(f0.z); t.h[3] = f32_to_bf16(f0.w);
                t.h[4] = f32_to_bf16(f1.x); t.h[5] = f32_to_bf16(f1.y);
                t.h[6] = f32_to_bf16(f1.z); t.h[7] = f32_to_bf16(f1.w);
            } else {
                t.q = make_uint4(0u, 0u, 0u, 0u);
            }
            *(uint4*)&Asb[lrow * AP + lchunk * 8] = t.q;
        } else {
            u32 lds  = (u32)(uintptr_t)&Asb[lrow * AP + lchunk * 8];
            u32 voff = (u32)(lrow * K + lchunk * 8) * 2u;
            u64 base = (u64)(uintptr_t)((const u16*)Av + (size_t)mbase * K + k0);
            async_b128(lds, voff, base);
        }
    };

    union Frag { uint4 u[2]; v16bf v; };

    stage(0, 0);
    for (int ks = 0; ks < S; ++ks) {
        wait_async0();                     // own tile-ks DMAs landed (in-order)
        __syncthreads();                   // everyone's landed; prev buffer free
        if (ks + 1 < S) stage(ks + 1, (ks + 1) & 1);

        const u16* Asb = As + (ks & 1) * ASZ;
        const u16* Bsb = Bs + (ks & 1) * BSZ;

        // Batch all fragment loads first (compiler spreads dscnt waits),
        // then issue the 8 WMMAs.
        Frag af;
        {
            const uint4* rowA = (const uint4*)&Asb[(mstrip * 16 + l15) * AP];
            af.u[0] = rowA[half];
            af.u[1] = rowA[2 + half];
        }
        Frag bfv[8];
        #pragma unroll
        for (int c = 0; c < 8; ++c) {
            const uint4* rowB = (const uint4*)&Bsb[(nbase + c * 16 + l15) * AP];
            bfv[c].u[0] = rowB[half];
            bfv[c].u[1] = rowB[2 + half];
        }
        #pragma unroll
        for (int c = 0; c < 8; ++c) {
            acc[c] = __builtin_amdgcn_wmma_f32_16x16x32_bf16(
                false, af.v, false, bfv[c].v, (short)0, acc[c], false, false);
        }
    }

    // ---- epilogue: C layout VGPR r -> M = r + 8*half, lane l15 -> N ----
    #pragma unroll
    for (int c = 0; c < 8; ++c) {
        const int n = nbase + c * 16 + l15;
        const float bv = ADD_BIAS ? bias[n] : 0.0f;
        #pragma unroll
        for (int r = 0; r < 8; ++r) {
            int m = mbase + mstrip * 16 + r + half * 8;
            if (m < M) {
                float v = acc[c][r] + bv;
                if (OUT_BF16) ((u16*)Out)[(size_t)m * HF + n] = f32_to_bf16(v);
                else          ((float*)Out)[(size_t)m * HF + n] = v;
            }
        }
    }
}

// ---------------- GCN scatter pieces ----------------

// s[i][f] = t[i][f] * dinv[i]^2   (self-loop term; also initializes s). 4-wide.
__global__ void self_loop_init(const float* __restrict__ t, const float* __restrict__ dinv,
                               float* __restrict__ s, int n4) {
    int i = blockIdx.x * blockDim.x + threadIdx.x;
    if (i < n4) {
        float d = dinv[i >> 6];            // (i*4)>>8
        d *= d;
        float4 v = ((const float4*)t)[i];
        v.x *= d; v.y *= d; v.z *= d; v.w *= d;
        ((float4*)s)[i] = v;
    }
}

// one wave per edge: s[dst] += t[src] * dinv[src]*dinv[dst]
__global__ __launch_bounds__(256) void scatter_edges(
    const int* __restrict__ src, const int* __restrict__ dst,
    const float* __restrict__ dinv, const float* __restrict__ t,
    float* __restrict__ s, int e)
{
    int w = blockIdx.x * 8 + (threadIdx.x >> 5);
    int lane = threadIdx.x & 31;
    if (w >= e) return;
    int u = src[w], v = dst[w];
    float norm = dinv[u] * dinv[v];
    const float* trow = t + (size_t)u * HF;
    float* srow = s + (size_t)v * HF;
    #pragma unroll
    for (int i = 0; i < 8; ++i) {
        int f = lane + 32 * i;
        atomicAdd(&srow[f], trow[f] * norm);
    }
}

// h[i][f] = bf16(relu(s[i][f] + b[f]))   8 elements per thread, packed b128 store
__global__ void bias_relu_bf16(const float* __restrict__ s, const float* __restrict__ b,
                               u16* __restrict__ h, int n8) {
    int i = blockIdx.x * blockDim.x + threadIdx.x;
    if (i < n8) {
        int e0 = i * 8;
        int f0 = e0 & (HF - 1);
        const float4* s4 = (const float4*)(s + e0);
        const float4* b4 = (const float4*)(b + f0);
        float4 a0 = s4[0], a1 = s4[1];
        float4 c0 = b4[0], c1 = b4[1];
        union { u16 h[8]; uint4 q; } o;
        o.h[0] = f32_to_bf16(fmaxf(a0.x + c0.x, 0.0f));
        o.h[1] = f32_to_bf16(fmaxf(a0.y + c0.y, 0.0f));
        o.h[2] = f32_to_bf16(fmaxf(a0.z + c0.z, 0.0f));
        o.h[3] = f32_to_bf16(fmaxf(a0.w + c0.w, 0.0f));
        o.h[4] = f32_to_bf16(fmaxf(a1.x + c1.x, 0.0f));
        o.h[5] = f32_to_bf16(fmaxf(a1.y + c1.y, 0.0f));
        o.h[6] = f32_to_bf16(fmaxf(a1.z + c1.z, 0.0f));
        o.h[7] = f32_to_bf16(fmaxf(a1.w + c1.w, 0.0f));
        *(uint4*)(h + e0) = o.q;
    }
}

// ---------------- pooling + classifier ----------------

#define POOL_NODES 100   // batch is sorted -> segmented accumulation, few atomics

__global__ __launch_bounds__(256) void pool_accum(
    const u16* __restrict__ h, const int* __restrict__ batch,
    float* __restrict__ sums, float* __restrict__ cnt)
{
    int f  = threadIdx.x;                  // 256 = HF
    int n0 = blockIdx.x * POOL_NODES;
    int n1 = n0 + POOL_NODES; if (n1 > NN) n1 = NN;
    int cur = batch[n0];
    float acc = 0.0f, c = 0.0f;
    for (int node = n0; node < n1; ++node) {
        int g = batch[node];               // uniform scalar load
        if (g != cur) {
            atomicAdd(&sums[cur * HF + f], acc);
            if (f == 0) atomicAdd(&cnt[cur], c);
            acc = 0.0f; c = 0.0f; cur = g;
        }
        acc += bf16_to_f32(h[(size_t)node * HF + f]);
        c += 1.0f;
    }
    atomicAdd(&sums[cur * HF + f], acc);
    if (f == 0) atomicAdd(&cnt[cur], c);
}

__global__ void pool_div(float* __restrict__ sums, const float* __restrict__ cnt, int n) {
    int i = blockIdx.x * blockDim.x + threadIdx.x;
    if (i < n) sums[i] = sums[i] / fmaxf(cnt[i >> 8], 1.0f);
}

__global__ void fc1_relu(const float* __restrict__ pooled, const float* __restrict__ W,
                         const float* __restrict__ b, float* __restrict__ z) {
    int i = blockIdx.x * blockDim.x + threadIdx.x;  // NG*128
    if (i < NG * 128) {
        int g = i >> 7, j = i & 127;
        float acc = b[j];
        for (int f = 0; f < HF; ++f) acc += pooled[g * HF + f] * W[f * 128 + j];
        z[i] = fmaxf(acc, 0.0f);
    }
}

__global__ void fc2(const float* __restrict__ z, const float* __restrict__ W,
                    const float* __restrict__ b, float* __restrict__ out) {
    int i = blockIdx.x * blockDim.x + threadIdx.x;  // NG*2
    if (i < NG * 2) {
        int g = i >> 1, c = i & 1;
        float acc = b[c];
        for (int j = 0; j < 128; ++j) acc += z[g * 128 + j] * W[j * 2 + c];
        out[i] = acc;
    }
}

// ---------------- launch ----------------

extern "C" void kernel_launch(void* const* d_in, const int* in_sizes, int n_in,
                              void* d_out, int out_size, void* d_ws, size_t ws_size,
                              hipStream_t stream) {
    (void)in_sizes; (void)n_in; (void)out_size; (void)ws_size;

    const float* x    = (const float*)d_in[0];
    const int*   ei   = (const int*)d_in[1];     // [2, E]: src row then dst row
    const int*   batch= (const int*)d_in[2];
    const float* Wp   = (const float*)d_in[3];
    const float* bp   = (const float*)d_in[4];
    const float* Wl[3]= {(const float*)d_in[5], (const float*)d_in[7], (const float*)d_in[9]};
    const float* bl[3]= {(const float*)d_in[6], (const float*)d_in[8], (const float*)d_in[10]};
    const float* Wc1  = (const float*)d_in[11];
    const float* bc1  = (const float*)d_in[12];
    const float* Wc2  = (const float*)d_in[13];
    const float* bc2  = (const float*)d_in[14];
    float* out = (float*)d_out;

    char* ws = (char*)d_ws;
    size_t off = 0;
    auto alloc = [&](size_t bytes) -> void* {
        void* p = ws + off;
        off = (off + bytes + 255) & ~(size_t)255;
        return p;
    };

    float* dinv = (float*)alloc((size_t)NN * 4);
    u16*   hbuf = (u16*)  alloc((size_t)NN * HF * 2);
    float* tbuf = (float*)alloc((size_t)NN * HF * 4);   // directly after hbuf: safe over-read pad
    float* sbuf = (float*)alloc((size_t)NN * HF * 4);
    u16*   Wtp  = (u16*)  alloc((size_t)DIN * HF * 2);
    u16*   Wt[3];
    Wt[0] = (u16*)alloc((size_t)HF * HF * 2);
    Wt[1] = (u16*)alloc((size_t)HF * HF * 2);
    Wt[2] = (u16*)alloc((size_t)HF * HF * 2);
    float* psum = (float*)alloc((size_t)(NG * HF + NG) * 4);
    float* pcnt = psum + NG * HF;
    float* z1   = (float*)alloc((size_t)NG * 128 * 4);

    const int* esrc = ei;
    const int* edst = ei + NE;

    // degrees (self-loop => init 1.0) -> dinv = deg^-1/2
    fill_f32<<<(NN + 255) / 256, 256, 0, stream>>>(dinv, 1.0f, NN);
    deg_accum<<<(NE + 255) / 256, 256, 0, stream>>>(edst, dinv, NE);
    deg_to_dinv<<<(NN + 255) / 256, 256, 0, stream>>>(dinv, NN);

    // weight convert + transpose to bf16 [N][K]
    cvt_transpose<<<(DIN * HF + 255) / 256, 256, 0, stream>>>(Wp, Wtp, DIN, HF);
    for (int l = 0; l < 3; ++l)
        cvt_transpose<<<(HF * HF + 255) / 256, 256, 0, stream>>>(Wl[l], Wt[l], HF, HF);

    const int gemm_blocks = (NN + 63) / 64;

    // projection: h = bf16(x @ Wp + bp)
    gemm_wmma<true, true, true><<<gemm_blocks, 256, 0, stream>>>(
        x, Wtp, bp, hbuf, NN, DIN);

    const int n4 = NN * HF / 4, n8 = NN * HF / 8;
    for (int l = 0; l < 3; ++l) {
        // t = h @ Wl   (fp32 out)
        gemm_wmma<false, false, false><<<gemm_blocks, 256, 0, stream>>>(
            hbuf, Wt[l], nullptr, tbuf, NN, HF);
        // s = t * dinv^2  (self-loop, also init)
        self_loop_init<<<(n4 + 255) / 256, 256, 0, stream>>>(tbuf, dinv, sbuf, n4);
        // s[dst] += t[src] * dinv[src]*dinv[dst]
        scatter_edges<<<(NE + 7) / 8, 256, 0, stream>>>(esrc, edst, dinv, tbuf, sbuf, NE);
        // h = bf16(relu(s + b))
        bias_relu_bf16<<<(n8 + 255) / 256, 256, 0, stream>>>(sbuf, bl[l], hbuf, n8);
    }

    // global mean pool (segmented: batch is sorted)
    hipMemsetAsync(psum, 0, (size_t)(NG * HF + NG) * 4, stream);
    pool_accum<<<(NN + POOL_NODES - 1) / POOL_NODES, HF, 0, stream>>>(hbuf, batch, psum, pcnt);
    pool_div<<<(NG * HF + 255) / 256, 256, 0, stream>>>(psum, pcnt, NG * HF);

    // classifier
    fc1_relu<<<(NG * 128 + 255) / 256, 256, 0, stream>>>(psum, Wc1, bc1, z1);
    fc2<<<1, 128, 0, stream>>>(z1, Wc2, bc2, out);
}
